// BasicSelfAttention_14525579395689
// MI455X (gfx1250) — compile-verified, matrix operand order checked by reading, not run on hardware
//
#include <hip/hip_runtime.h>
#include <hip/hip_bf16.h>

typedef __attribute__((ext_vector_type(16))) __bf16 v16bf;
typedef __attribute__((ext_vector_type(8)))  __bf16 v8bf;
typedef __attribute__((ext_vector_type(8)))  float  v8f;
typedef __attribute__((ext_vector_type(4)))  int    v4i;

#define NSEQ 4096
#define DDIM 512
#define BATCH 4
#define BM 32          // query rows per workgroup
#define BN 64          // key rows per iteration
#define QPITCH 520     // bf16 elements (pad to break bank alignment)
#define KPITCH 520
#define SPITCH 68      // f32 elements
#define PPITCH 72      // bf16 elements
#define VPITCH 72      // bf16 elements (Vt tile rows = d index)

#if defined(__has_builtin)
#if __has_builtin(__builtin_amdgcn_global_load_async_to_lds_b128) && \
    __has_builtin(__builtin_amdgcn_s_wait_asynccnt)
#define USE_ASYNC_COPY 1
#endif
#endif

typedef __attribute__((address_space(1))) v4i gl_v4i;   // global int4
typedef __attribute__((address_space(3))) v4i ds_v4i;   // LDS int4

union ABFrag { v16bf v; v8bf h[2]; };

// ---------------- prepass: fp32 X -> bf16 row-major Xb and transposed Xt ----
__global__ __launch_bounds__(256)
void cvt_transpose_kernel(const float* __restrict__ X,
                          __bf16* __restrict__ Xb, __bf16* __restrict__ Xt) {
  __shared__ __bf16 tile[32][33];
  int b  = blockIdx.z;
  int n0 = blockIdx.x * 32;
  int d0 = blockIdx.y * 32;
  int tx = threadIdx.x;          // 0..31
  int ty = threadIdx.y;          // 0..7
  const float* Xp = X + (size_t)b * NSEQ * DDIM;
  #pragma unroll
  for (int k = 0; k < 32; k += 8) {
    float v = Xp[(size_t)(n0 + ty + k) * DDIM + d0 + tx];
    __bf16 w = (__bf16)v;
    Xb[((size_t)b * NSEQ + n0 + ty + k) * DDIM + d0 + tx] = w;
    tile[ty + k][tx] = w;
  }
  __syncthreads();
  #pragma unroll
  for (int k = 0; k < 32; k += 8) {
    Xt[((size_t)b * DDIM + d0 + ty + k) * NSEQ + n0 + tx] = tile[tx][ty + k];
  }
}

__device__ __forceinline__ void cp16_g2l(__bf16* dst, const __bf16* src) {
#if USE_ASYNC_COPY
  __builtin_amdgcn_global_load_async_to_lds_b128(
      (gl_v4i*)(src), (ds_v4i*)(dst), 0, 0);
#else
  *(v8bf*)dst = *(const v8bf*)src;
#endif
}

// ---------------- flash attention main kernel ------------------------------
__global__ __launch_bounds__(256)
void flash_attn_kernel(const __bf16* __restrict__ Xb,
                       const __bf16* __restrict__ Xt,
                       float* __restrict__ out) {
  extern __shared__ char smem_raw[];
  __bf16* Qs   = (__bf16*)smem_raw;                 // BM x QPITCH
  __bf16* Ks   = Qs + BM * QPITCH;                  // BN x KPITCH
  float*  Ss   = (float*)(Ks + BN * KPITCH);        // BM x SPITCH
  __bf16* Ps   = (__bf16*)(Ss + BM * SPITCH);       // BM x PPITCH
  float*  mrun = (float*)(Ps + BM * PPITCH);        // BM
  float*  lrun = mrun + BM;                         // BM
  float*  alph = lrun + BM;                         // BM
  __bf16* Vs   = (__bf16*)(alph + BM);              // DDIM x VPITCH (transposed V)

  const int tid  = threadIdx.x;
  const int wave = tid >> 5;
  const int lane = tid & 31;
  const int n15  = lane & 15;
  const int hi   = lane >> 4;

  const int b     = blockIdx.x >> 7;       // 128 row-blocks per batch
  const int mbase = (blockIdx.x & 127) * BM;

  const __bf16* Xbb = Xb + (size_t)b * NSEQ * DDIM;
  const __bf16* Xtb = Xt + (size_t)b * DDIM * NSEQ;

  // cooperative load of Q tile (BM x DDIM bf16) into LDS
  for (int c = tid; c < BM * (DDIM / 8); c += 256) {
    int row = c >> 6;                      // DDIM/8 == 64 chunks per row
    int col = (c & 63) * 8;
    cp16_g2l(Qs + row * QPITCH + col,
             Xbb + (size_t)(mbase + row) * DDIM + col);
  }
  if (tid < BM) { mrun[tid] = -__builtin_inff(); lrun[tid] = 0.f; }

  const int mi = wave & 1;     // 16-row strip within the 32-row tile
  const int nt = wave >> 1;    // S-tile column index (QK phase), 0..3
  const int jc = wave >> 1;    // 128-wide D chunk (PV phase), 0..3

  v8f o[8];
  #pragma unroll
  for (int jt = 0; jt < 8; ++jt)
    #pragma unroll
    for (int r = 0; r < 8; ++r) o[jt][r] = 0.f;

  for (int t = 0; t < NSEQ / BN; ++t) {
    // ---- cooperative load of K tile (BN x DDIM) + Vt tile (DDIM x BN) ----
    for (int c = tid; c < BN * (DDIM / 8); c += 256) {
      int row = c >> 6;
      int col = (c & 63) * 8;
      cp16_g2l(Ks + row * KPITCH + col,
               Xbb + (size_t)(t * BN + row) * DDIM + col);
    }
    for (int c = tid; c < DDIM * (BN / 8); c += 256) {
      int row = c >> 3;                    // BN/8 == 8 chunks per row
      int col = (c & 7) * 8;
      cp16_g2l(Vs + row * VPITCH + col,
               Xtb + (size_t)row * NSEQ + t * BN + col);
    }
#if USE_ASYNC_COPY
    __builtin_amdgcn_s_wait_asynccnt(0);
#endif
    __syncthreads();

    // ---- S = Q K^T : each wave computes one 16x16 tile, K-dim 512 ----
    {
      v8f s;
      #pragma unroll
      for (int r = 0; r < 8; ++r) s[r] = 0.f;
      #pragma unroll 4
      for (int kc = 0; kc < DDIM / 32; ++kc) {
        ABFrag a, kb;
        const __bf16* ap = Qs + (mi * 16 + n15) * QPITCH + kc * 32 + hi * 8;
        a.h[0] = *(const v8bf*)(ap);
        a.h[1] = *(const v8bf*)(ap + 16);
        const __bf16* bp = Ks + (nt * 16 + n15) * KPITCH + kc * 32 + hi * 8;
        kb.h[0] = *(const v8bf*)(bp);
        kb.h[1] = *(const v8bf*)(bp + 16);
        s = __builtin_amdgcn_wmma_f32_16x16x32_bf16(
            false, a.v, false, kb.v, (short)0, s, false, false);
      }
      #pragma unroll
      for (int r = 0; r < 8; ++r)
        Ss[(mi * 16 + hi * 8 + r) * SPITCH + nt * 16 + n15] = s[r];
    }
    __syncthreads();

    // ---- online softmax: each wave owns 4 rows (8 lanes per row) ----
    {
      int row = wave * 4 + (lane >> 3);
      int seg = lane & 7;
      const float4* sp = (const float4*)(Ss + row * SPITCH + seg * 8);
      float4 v0 = sp[0], v1 = sp[1];
      float mx = fmaxf(fmaxf(fmaxf(v0.x, v0.y), fmaxf(v0.z, v0.w)),
                       fmaxf(fmaxf(v1.x, v1.y), fmaxf(v1.z, v1.w)));
      mx = fmaxf(mx, __shfl_xor(mx, 1, 8));
      mx = fmaxf(mx, __shfl_xor(mx, 2, 8));
      mx = fmaxf(mx, __shfl_xor(mx, 4, 8));
      float mold = mrun[row];
      float mnew = fmaxf(mold, mx);
      float al   = __expf(mold - mnew);
      float p[8];
      p[0] = __expf(v0.x - mnew); p[1] = __expf(v0.y - mnew);
      p[2] = __expf(v0.z - mnew); p[3] = __expf(v0.w - mnew);
      p[4] = __expf(v1.x - mnew); p[5] = __expf(v1.y - mnew);
      p[6] = __expf(v1.z - mnew); p[7] = __expf(v1.w - mnew);
      float sum = p[0] + p[1] + p[2] + p[3] + p[4] + p[5] + p[6] + p[7];
      sum += __shfl_xor(sum, 1, 8);
      sum += __shfl_xor(sum, 2, 8);
      sum += __shfl_xor(sum, 4, 8);
      if (seg == 0) {
        mrun[row] = mnew;
        lrun[row] = lrun[row] * al + sum;
        alph[row] = al;
      }
      v8bf pk;
      #pragma unroll
      for (int i = 0; i < 8; ++i) pk[i] = (__bf16)p[i];
      *(v8bf*)(Ps + row * PPITCH + seg * 8) = pk;
    }
    __syncthreads();

    // ---- O = diag(alpha) O + P V : wave owns 16 rows x 128 cols ----
    {
      const float* ap0 = alph + mi * 16 + hi * 8;
      float4 a0 = *(const float4*)(ap0);
      float4 a1 = *(const float4*)(ap0 + 4);
      float av[8] = {a0.x, a0.y, a0.z, a0.w, a1.x, a1.y, a1.z, a1.w};
      #pragma unroll
      for (int jt = 0; jt < 8; ++jt)
        #pragma unroll
        for (int r = 0; r < 8; ++r) o[jt][r] *= av[r];

      ABFrag pa[2];
      #pragma unroll
      for (int kc = 0; kc < 2; ++kc) {
        const __bf16* pp = Ps + (mi * 16 + n15) * PPITCH + kc * 32 + hi * 8;
        pa[kc].h[0] = *(const v8bf*)(pp);
        pa[kc].h[1] = *(const v8bf*)(pp + 16);
      }

      // 16 WMMA steps, kc-outer / jt-inner so consecutive WMMAs hit
      // different accumulators; V fragments prefetched 2 steps ahead in a
      // 4-deep rotating buffer -> partial s_wait_dscnt, loads stay in flight.
      const __bf16* vbase = Vs + (jc * 128 + n15) * VPITCH + hi * 8;
      ABFrag vb[4];
      #pragma unroll
      for (int s = 0; s < 2; ++s) {        // preload steps 0,1
        int kc = s >> 3, jt = s & 7;
        const __bf16* vp = vbase + jt * 16 * VPITCH + kc * 32;
        vb[s].h[0] = *(const v8bf*)(vp);
        vb[s].h[1] = *(const v8bf*)(vp + 16);
      }
      #pragma unroll
      for (int s = 0; s < 16; ++s) {
        if (s < 14) {
          int s2 = s + 2;
          int kc2 = s2 >> 3, jt2 = s2 & 7;
          const __bf16* vp = vbase + jt2 * 16 * VPITCH + kc2 * 32;
          vb[s2 & 3].h[0] = *(const v8bf*)(vp);
          vb[s2 & 3].h[1] = *(const v8bf*)(vp + 16);
        }
        int kc = s >> 3, jt = s & 7;
        o[jt] = __builtin_amdgcn_wmma_f32_16x16x32_bf16(
            false, pa[kc].v, false, vb[s & 3].v, (short)0, o[jt], false, false);
      }
    }
    __syncthreads();
  }

  // ---- epilogue: O /= l, store fp32 ----
  {
    const float* lp = lrun + mi * 16 + hi * 8;
    float4 l0 = *(const float4*)(lp);
    float4 l1 = *(const float4*)(lp + 4);
    float lv[8] = {l0.x, l0.y, l0.z, l0.w, l1.x, l1.y, l1.z, l1.w};
    #pragma unroll
    for (int r = 0; r < 8; ++r) lv[r] = 1.f / lv[r];
    #pragma unroll
    for (int jt = 0; jt < 8; ++jt) {
      int j = jc * 128 + jt * 16 + n15;
      #pragma unroll
      for (int r = 0; r < 8; ++r) {
        int m = mbase + mi * 16 + hi * 8 + r;
        out[((size_t)b * NSEQ + m) * DDIM + j] = o[jt][r] * lv[r];
      }
    }
  }
}

extern "C" void kernel_launch(void* const* d_in, const int* in_sizes, int n_in,
                              void* d_out, int out_size, void* d_ws, size_t ws_size,
                              hipStream_t stream) {
  (void)in_sizes; (void)n_in; (void)out_size; (void)ws_size;
  const float* X = (const float*)d_in[0];
  float* out = (float*)d_out;

  __bf16* Xb = (__bf16*)d_ws;                              // [4][4096][512] bf16
  __bf16* Xt = Xb + (size_t)BATCH * NSEQ * DDIM;           // [4][512][4096] bf16

  dim3 gtr(NSEQ / 32, DDIM / 32, BATCH), btr(32, 8, 1);
  cvt_transpose_kernel<<<gtr, btr, 0, stream>>>(X, Xb, Xt);

  size_t smem = (size_t)BM * QPITCH * 2 + (size_t)BN * KPITCH * 2 +
                (size_t)BM * SPITCH * 4 + (size_t)BM * PPITCH * 2 +
                3 * BM * 4 +
                (size_t)DDIM * VPITCH * 2;   // ~183 KB dynamic LDS (<= 320 KB/WGP)
  flash_attn_kernel<<<dim3(BATCH * (NSEQ / BM)), dim3(256), smem, stream>>>(
      Xb, Xt, out);
}
